// Net_89343909691631
// MI455X (gfx1250) — matrix operand order repaired and checked by reading, not compiled, not run on hardware
//
#include <hip/hip_runtime.h>
#include <math.h>

typedef __bf16 v16bf __attribute__((ext_vector_type(16)));
typedef float v8f __attribute__((ext_vector_type(8)));
typedef unsigned short ushort8_t __attribute__((ext_vector_type(8)));
typedef unsigned short ushort4_t __attribute__((ext_vector_type(4)));
typedef float float4_t __attribute__((ext_vector_type(4)));

#define DD 512
#define HH 512
#define OO 512
#define EE 8
#define TM 64           // token rows per block
#define NTHREADS 256    // 8 waves -> 2 waves/SIMD -> full 256-VGPR budget (no spills)
#define SMEM_STRIDE 520 // ushorts per x/h row: 1040B = 65*16 -> 16B-aligned rows
#define BSTRIDE 40      // ushorts per staged-B row: 80B -> 16B-aligned, conflict-free b128

__device__ __forceinline__ unsigned short f2bf(float f) {
  unsigned int u = __builtin_bit_cast(unsigned int, f);
  unsigned int r = (u + 0x7FFFu + ((u >> 16) & 1u)) >> 16;  // RNE
  return (unsigned short)r;
}

__device__ __forceinline__ unsigned int lds_addr(const void* p) {
  // raw LDS byte offset (addrspace(3)) from a generic pointer to __shared__
  return (unsigned int)(size_t)(__attribute__((address_space(3))) const void*)p;
}

__device__ __forceinline__ void wait_async0() {
  asm volatile("s_wait_asynccnt 0x0" ::: "memory");
}

// Stage 2 consecutive B rows (2 x 64B) this thread owns: 8 async b128 copies.
// IOFFSET is added to BOTH lds and global addresses (ISA 15.18.3 async pseudocode),
// and intra-row deltas match on both sides, so chunk offsets fold into immediates.
__device__ __forceinline__ void stage2rows(unsigned int lds0, const unsigned short* g0) {
  const unsigned short* g1 = g0 + DD;        // next row: global +1024B
  unsigned int lds1 = lds0 + BSTRIDE * 2;    // next row: LDS +80B
  asm volatile(
      "global_load_async_to_lds_b128 %0, %2, off\n\t"
      "global_load_async_to_lds_b128 %0, %2, off offset:16\n\t"
      "global_load_async_to_lds_b128 %0, %2, off offset:32\n\t"
      "global_load_async_to_lds_b128 %0, %2, off offset:48\n\t"
      "global_load_async_to_lds_b128 %1, %3, off\n\t"
      "global_load_async_to_lds_b128 %1, %3, off offset:16\n\t"
      "global_load_async_to_lds_b128 %1, %3, off offset:32\n\t"
      "global_load_async_to_lds_b128 %1, %3, off offset:48"
      :: "v"(lds0), "v"(lds1), "v"(g0), "v"(g1) : "memory");
}

union FragU { v16bf v; ushort8_t h[2]; };

// A fragment (16x32 bf16) from LDS x/h tile.
__device__ __forceinline__ v16bf loadA(const unsigned short* sm, int row0, int k0, int lane) {
  int m = lane & 15;
  int koff = k0 + ((lane >> 4) << 3);
  const unsigned short* p = sm + (row0 + m) * SMEM_STRIDE + koff;
  FragU u;
  u.h[0] = *(const ushort8_t*)(p);
  u.h[1] = *(const ushort8_t*)(p + 16);
  return u.v;
}

// B fragment (32x16 bf16) from staged LDS tile [n][0..31] with BSTRIDE.
__device__ __forceinline__ v16bf loadBs(const unsigned short* bbuf, int n0, int lane) {
  int col = n0 + (lane & 15);
  int koff = (lane >> 4) << 3;
  const unsigned short* p = bbuf + col * BSTRIDE + koff;
  FragU u;
  u.h[0] = *(const ushort8_t*)(p);
  u.h[1] = *(const ushort8_t*)(p + 16);
  return u.v;
}

// ---------------- weight transpose + fp32->bf16 convert ----------------
// src: [E][R][C] fp32 -> dst: [E][C][R] bf16
__global__ __launch_bounds__(256) void convT(const float* __restrict__ src,
                                             unsigned short* __restrict__ dst,
                                             int R, int C) {
  __shared__ float tile[32][33];
  const int e = blockIdx.z;
  const float* s = src + (size_t)e * R * C;
  unsigned short* d = dst + (size_t)e * R * C;
  const int c0 = blockIdx.x * 32, r0 = blockIdx.y * 32;
  const int tx = threadIdx.x & 31, ty = threadIdx.x >> 5;
#pragma unroll
  for (int i = 0; i < 4; ++i)
    tile[ty + i * 8][tx] = s[(size_t)(r0 + ty + i * 8) * C + c0 + tx];
  __syncthreads();
#pragma unroll
  for (int i = 0; i < 4; ++i)
    d[(size_t)(c0 + ty + i * 8) * R + r0 + tx] = f2bf(tile[tx][ty + i * 8]);
}

// ---------------- gating: softmax -> top-2 -> L1 norm -> route ----------------
__global__ __launch_bounds__(256) void moe_gate(const float* __restrict__ x,
                                                const float* __restrict__ Wg, int N,
                                                int* __restrict__ counts,
                                                int* __restrict__ tok,
                                                float* __restrict__ wt) {
  __shared__ float wg_s[DD * EE];
  const int tid = threadIdx.x;
  for (int i = tid; i < DD * EE; i += 256) wg_s[i] = Wg[i];
  __syncthreads();
  const int lane = tid & 31, wave = tid >> 5;
  const int t = blockIdx.x * 8 + wave;
  if (t >= N) return;
  float acc[EE] = {};
  const float* xr = x + (size_t)t * DD;
#pragma unroll
  for (int j = 0; j < DD / 32; ++j) {
    float xv = xr[j * 32 + lane];
    const float* w = &wg_s[(j * 32 + lane) * EE];
#pragma unroll
    for (int e = 0; e < EE; ++e) acc[e] += xv * w[e];
  }
#pragma unroll
  for (int m = 16; m >= 1; m >>= 1)
#pragma unroll
    for (int e = 0; e < EE; ++e) acc[e] += __shfl_xor(acc[e], m, 32);
  if (lane == 0) {
    float mx = acc[0];
#pragma unroll
    for (int e = 1; e < EE; ++e) mx = fmaxf(mx, acc[e]);
    float p[EE];
#pragma unroll
    for (int e = 0; e < EE; ++e) p[e] = __expf(acc[e] - mx);
    int i1 = 0;
#pragma unroll
    for (int e = 1; e < EE; ++e) if (p[e] > p[i1]) i1 = e;
    int i2 = (i1 == 0) ? 1 : 0;
#pragma unroll
    for (int e = 0; e < EE; ++e) if (e != i1 && p[e] > p[i2]) i2 = e;
    float inv = 1.0f / (p[i1] + p[i2]);  // softmax denom cancels in L1 norm
    float w1 = p[i1] * inv, w2 = p[i2] * inv;
    int s1 = atomicAdd(&counts[i1], 1);
    tok[(size_t)i1 * N + s1] = t; wt[(size_t)i1 * N + s1] = w1;
    int s2 = atomicAdd(&counts[i2], 1);
    tok[(size_t)i2 * N + s2] = t; wt[(size_t)i2 * N + s2] = w2;
  }
}

// ---------------- fused expert FFN ----------------
__global__ __launch_bounds__(NTHREADS) void moe_ffn(const float* __restrict__ x,
                                                    const unsigned short* __restrict__ W1t,  // [E][H][D] bf16
                                                    const float* __restrict__ gamma,
                                                    const float* __restrict__ beta,
                                                    const unsigned short* __restrict__ W2t,  // [E][O][H] bf16
                                                    const int* __restrict__ counts,
                                                    const int* __restrict__ tok,
                                                    const float* __restrict__ wt,
                                                    float* __restrict__ out, int N) {
  __shared__ __align__(16) unsigned short smem[TM * SMEM_STRIDE];  // 65 KB: x tile, then h tile
  __shared__ __align__(16) unsigned short bt[2][DD * BSTRIDE];     // 2 x 40 KB staged B tiles
  __shared__ float red[TM][4][2];
  __shared__ float muS[TM], rsS[TM];
  __shared__ int stok[TM];
  __shared__ float swt[TM];

  const int e = blockIdx.y;
  const int cnt = counts[e];
  const int tile0 = blockIdx.x * TM;
  if (tile0 >= cnt) return;
  const int rows = min(TM, cnt - tile0);

  const int tid = threadIdx.x;
  const int lane = tid & 31, wid = tid >> 5;
  const int rg = wid & 1;   // 2 row groups x 32 rows
  const int cg = wid >> 1;  // 4 col groups x 128 cols

  // per-thread staging bases: 2 consecutive B rows each
  const int nrow = tid * 2;
  const unsigned int lbuf0 = lds_addr(&bt[0][nrow * BSTRIDE]);
  const unsigned int lbuf1 = lds_addr(&bt[1][nrow * BSTRIDE]);

  if (tid < TM) {
    int tkn = (tid < rows) ? tok[(size_t)e * N + tile0 + tid] : -1;
    stok[tid] = tkn;
    swt[tid] = (tid < rows) ? wt[(size_t)e * N + tile0 + tid] : 0.0f;
  }
  __syncthreads();

  // ---- gather x rows -> LDS as bf16 (zero-pad invalid rows) ----
#pragma unroll
  for (int rr = 0; rr < TM / 8; ++rr) {
    int r = wid * (TM / 8) + rr;
    int tkn = stok[r];
#pragma unroll
    for (int pass = 0; pass < 4; ++pass) {
      int c = pass * 128 + lane * 4;
      float4_t v = {0.f, 0.f, 0.f, 0.f};
      if (tkn >= 0) v = *(const float4_t*)(x + (size_t)tkn * DD + c);
      ushort4_t b;
      b.x = f2bf(v.x); b.y = f2bf(v.y); b.z = f2bf(v.z); b.w = f2bf(v.w);
      *(ushort4_t*)(&smem[r * SMEM_STRIDE + c]) = b;
    }
  }

  const v8f vzero = {0.f, 0.f, 0.f, 0.f, 0.f, 0.f, 0.f, 0.f};
  v8f c[2][8];
#pragma unroll
  for (int i = 0; i < 2; ++i)
#pragma unroll
    for (int j = 0; j < 8; ++j) c[i][j] = vzero;

  // ---- GEMM1: h = x @ W1[e], async double-buffered B staging ----
  const unsigned short* W1row = W1t + (size_t)e * HH * DD + (size_t)nrow * DD;
  stage2rows(lbuf0, W1row);
  for (int it = 0; it < DD / 32; ++it) {
    wait_async0();        // own async copies for bt[it&1] landed
    __syncthreads();      // everyone's copies landed; previous buffer free
    if (it + 1 < DD / 32)
      stage2rows((it & 1) ? lbuf0 : lbuf1, W1row + (it + 1) * 32);
    const unsigned short* bb = bt[it & 1];
    v16bf a0 = loadA(smem, rg * 32, it * 32, lane);
    v16bf a1 = loadA(smem, rg * 32 + 16, it * 32, lane);
#pragma unroll
    for (int j = 0; j < 8; ++j) {
      v16bf b = loadBs(bb, cg * 128 + j * 16, lane);
      c[0][j] = __builtin_amdgcn_wmma_f32_16x16x32_bf16(false, a0, false, b, (short)0, c[0][j], false, false);
      c[1][j] = __builtin_amdgcn_wmma_f32_16x16x32_bf16(false, a1, false, b, (short)0, c[1][j], false, false);
    }
  }

  // ---- LayerNorm stats ----
#pragma unroll
  for (int i = 0; i < 2; ++i)
#pragma unroll
    for (int r = 0; r < 8; ++r) {
      float s = 0.f, q = 0.f;
#pragma unroll
      for (int j = 0; j < 8; ++j) { float v = c[i][j][r]; s += v; q += v * v; }
#pragma unroll
      for (int m = 8; m >= 1; m >>= 1) { s += __shfl_xor(s, m, 32); q += __shfl_xor(q, m, 32); }
      if ((lane & 15) == 0) {
        int row = rg * 32 + i * 16 + ((lane >> 4) << 3) + r;
        red[row][cg][0] = s;
        red[row][cg][1] = q;
      }
    }
  __syncthreads();
  if (tid < TM) {
    float s = red[tid][0][0] + red[tid][1][0] + red[tid][2][0] + red[tid][3][0];
    float q = red[tid][0][1] + red[tid][1][1] + red[tid][2][1] + red[tid][3][1];
    float mu = s * (1.0f / (float)HH);
    float var = q * (1.0f / (float)HH) - mu * mu;
    muS[tid] = mu;
    rsS[tid] = rsqrtf(var + 1e-5f);
  }
  __syncthreads();

  // ---- normalize + exact GELU, write h back into smem as bf16 ----
  const float* ga = gamma + (size_t)e * HH;
  const float* be = beta + (size_t)e * HH;
#pragma unroll
  for (int i = 0; i < 2; ++i)
#pragma unroll
    for (int j = 0; j < 8; ++j) {
      int col = cg * 128 + j * 16 + (lane & 15);
      float g = ga[col], b = be[col];
#pragma unroll
      for (int r = 0; r < 8; ++r) {
        int row = rg * 32 + i * 16 + ((lane >> 4) << 3) + r;
        float v = (c[i][j][r] - muS[row]) * rsS[row] * g + b;
        v = 0.5f * v * (1.0f + erff(v * 0.70710678118f));
        smem[row * SMEM_STRIDE + col] = f2bf(v);
      }
    }
  __syncthreads();

  // ---- GEMM2: y = h @ W2[e] ----
#pragma unroll
  for (int i = 0; i < 2; ++i)
#pragma unroll
    for (int j = 0; j < 8; ++j) c[i][j] = vzero;
  const unsigned short* W2row = W2t + (size_t)e * OO * HH + (size_t)nrow * HH;
  stage2rows(lbuf0, W2row);
  for (int it = 0; it < HH / 32; ++it) {
    wait_async0();
    __syncthreads();
    if (it + 1 < HH / 32)
      stage2rows((it & 1) ? lbuf0 : lbuf1, W2row + (it + 1) * 32);
    const unsigned short* bb = bt[it & 1];
    v16bf a0 = loadA(smem, rg * 32, it * 32, lane);
    v16bf a1 = loadA(smem, rg * 32 + 16, it * 32, lane);
#pragma unroll
    for (int j = 0; j < 8; ++j) {
      v16bf b = loadBs(bb, cg * 128 + j * 16, lane);
      c[0][j] = __builtin_amdgcn_wmma_f32_16x16x32_bf16(false, a0, false, b, (short)0, c[0][j], false, false);
      c[1][j] = __builtin_amdgcn_wmma_f32_16x16x32_bf16(false, a1, false, b, (short)0, c[1][j], false, false);
    }
  }

  // ---- weighted scatter-add into out ----
#pragma unroll
  for (int i = 0; i < 2; ++i)
#pragma unroll
    for (int j = 0; j < 8; ++j) {
      int col = cg * 128 + j * 16 + (lane & 15);
#pragma unroll
      for (int r = 0; r < 8; ++r) {
        int row = rg * 32 + i * 16 + ((lane >> 4) << 3) + r;
        int tkn = stok[row];
        if (tkn >= 0) atomicAdd(&out[(size_t)tkn * OO + col], swt[row] * c[i][j][r]);
      }
    }
}

extern "C" void kernel_launch(void* const* d_in, const int* in_sizes, int n_in,
                              void* d_out, int out_size, void* d_ws, size_t ws_size,
                              hipStream_t stream) {
  (void)n_in; (void)ws_size;
  const float* x     = (const float*)d_in[0];
  const float* Wg    = (const float*)d_in[1];
  const float* W1    = (const float*)d_in[2];
  const float* gamma = (const float*)d_in[3];
  const float* beta  = (const float*)d_in[4];
  const float* W2    = (const float*)d_in[5];
  float* out = (float*)d_out;
  const int N = in_sizes[0] / DD;

  char* ws = (char*)d_ws;
  int* counts = (int*)ws;                                  // EE ints (padded to 256B)
  int* tokl   = (int*)(ws + 256);                          // EE*N ints
  float* wtl  = (float*)(ws + 256 + (size_t)EE * N * 4);   // EE*N floats
  unsigned short* W1t = (unsigned short*)(ws + 256 + (size_t)2 * EE * N * 4);
  unsigned short* W2t = W1t + (size_t)EE * HH * DD;

  hipMemsetAsync(counts, 0, EE * sizeof(int), stream);
  hipMemsetAsync(out, 0, (size_t)out_size * sizeof(float), stream);

  dim3 tg(HH / 32, DD / 32, EE);
  convT<<<tg, 256, 0, stream>>>(W1, W1t, DD, HH);
  convT<<<tg, 256, 0, stream>>>(W2, W2t, HH, OO);

  moe_gate<<<(N + 7) / 8, 256, 0, stream>>>(x, Wg, N, counts, tokl, wtl);

  dim3 fg((N + TM - 1) / TM, EE);
  moe_ffn<<<fg, NTHREADS, 0, stream>>>(x, W1t, gamma, beta, W2t, counts, tokl, wtl, out, N);
}
